// Decoders_48498770706735
// MI455X (gfx1250) — compile-verified, matrix operand order checked by reading, not run on hardware
//
#include <hip/hip_runtime.h>
#include <math.h>

typedef __attribute__((ext_vector_type(8)))  _Float16 v8h;
typedef __attribute__((ext_vector_type(16))) _Float16 v16h;
typedef __attribute__((ext_vector_type(8)))  float    v8f;

// LDS row stride (halves): >=128 cols, 272B rows keep 16B alignment, odd*8 banks.
#define S_LDS 136

// f16 weight workspace offsets (halves); all byte offsets are 16B aligned.
#define OFF_LIN0   0        // 64x32   = 2048
#define OFF_LIN1   2048     // 64x64   = 4096
#define OFF_COMB0  6144     // 128x96  = 12288
#define OFF_COMB1  18432    // 128x128 = 16384
#define OFF_CLIN0  34816    // 64x32   = 2048
#define OFF_CLIN1  36864    // 64x64   = 4096
// total 40960 halves = 80 KB

__global__ void cvt_f32_to_f16(const float* __restrict__ src, _Float16* __restrict__ dst, int n) {
    int i = blockIdx.x * blockDim.x + threadIdx.x;
    if (i < n) dst[i] = (_Float16)src[i];
}

// Load one 16-lane fragment slice: 8 halves at kbase, 8 at kbase+16 (ISA 16-bit A/B layout).
static __device__ inline v16h frag_from(const _Float16* base, int kbase) {
    v8h lo = *(const v8h*)(base + kbase);
    v8h hi = *(const v8h*)(base + kbase + 16);
    v16h r;
#pragma unroll
    for (int i = 0; i < 8; ++i) { r[i] = lo[i]; r[i + 8] = hi[i]; }
    return r;
}

// OUT[16 x OUT_DIM] = act(IN[16 x IN_DIM] @ W^T + bias), f16 in LDS -> f16 in LDS.
// W is row-major [OUT_DIM, IN_DIM] f16 in global (K contiguous == B-fragment layout).
template <int IN_DIM, int OUT_DIM, bool RELU>
static __device__ inline void mlp_layer(const _Float16* __restrict__ ldsIn,
                                        _Float16* __restrict__ ldsOut,
                                        const _Float16* __restrict__ w,
                                        const float* __restrict__ bias, int lane) {
    const int col   = lane & 15;
    const int kbase = (lane >> 4) * 8;
    const int mbase = (lane >> 4) * 8;
    const _Float16* arow = ldsIn + col * S_LDS;
#pragma unroll
    for (int nt = 0; nt < OUT_DIM / 16; ++nt) {
        const _Float16* brow = w + (size_t)(nt * 16 + col) * IN_DIM;
        v8f acc = {};
#pragma unroll
        for (int kt = 0; kt < IN_DIM / 32; ++kt) {
            v16h a = frag_from(arow + kt * 32, kbase);
            v16h b = frag_from(brow + kt * 32, kbase);
            acc = __builtin_amdgcn_wmma_f32_16x16x32_f16(false, a, false, b,
                                                         (short)0, acc, false, false);
        }
        float bv = bias[nt * 16 + col];
#pragma unroll
        for (int r = 0; r < 8; ++r) {
            float v = acc[r] + bv;
            if (RELU) v = fmaxf(v, 0.0f);
            ldsOut[(mbase + r) * S_LDS + nt * 16 + col] = (_Float16)v;
        }
    }
}

// Bilinear sample (border clamp, align_corners) of 16 channels [chbase..chbase+15].
static __device__ inline void sample_plane_acc(const float* __restrict__ plane, int R,
                                               float a, float b, int chbase, float* acc) {
    float fx = (a + 1.0f) * 0.5f * (float)(R - 1);
    float fy = (b + 1.0f) * 0.5f * (float)(R - 1);
    float x0f = floorf(fx), y0f = floorf(fy);
    float wx = fx - x0f, wy = fy - y0f;
    int x0 = (int)x0f; x0 = x0 < 0 ? 0 : (x0 > R - 1 ? R - 1 : x0);
    int x1 = x0 + 1 > R - 1 ? R - 1 : x0 + 1;
    int y0 = (int)y0f; y0 = y0 < 0 ? 0 : (y0 > R - 1 ? R - 1 : y0);
    int y1 = y0 + 1 > R - 1 ? R - 1 : y0 + 1;
    float w00 = (1 - wx) * (1 - wy), w01 = wx * (1 - wy);
    float w10 = (1 - wx) * wy,       w11 = wx * wy;
    size_t pp = (size_t)R * R;
    const float* base = plane + (size_t)chbase * pp;
    int i00 = y0 * R + x0, i01 = y0 * R + x1, i10 = y1 * R + x0, i11 = y1 * R + x1;
#pragma unroll
    for (int c = 0; c < 16; ++c) {
        const float* pc = base + (size_t)c * pp;
        acc[c] = fmaf(w00, pc[i00],
                 fmaf(w01, pc[i01],
                 fmaf(w10, pc[i10],
                 fmaf(w11, pc[i11], acc[c]))));
    }
}

// Sum of 3 plane samples of one set -> f16 into LDS row (cols chbase..chbase+15 of dst).
static __device__ inline void sample_set(const float* pxy, const float* pxz, const float* pyz,
                                         int R, float X, float Y, float Z, int chbase,
                                         _Float16* dst) {
    float acc[16];
#pragma unroll
    for (int c = 0; c < 16; ++c) acc[c] = 0.0f;
    sample_plane_acc(pxy, R, X, Y, chbase, acc);
    sample_plane_acc(pxz, R, X, Z, chbase, acc);
    sample_plane_acc(pyz, R, Y, Z, chbase, acc);
#pragma unroll
    for (int c = 0; c < 16; ++c) dst[chbase + c] = (_Float16)acc[c];
}

__global__ __launch_bounds__(128) void triplane_decoder_kernel(
    const float* __restrict__ P,
    const float* __restrict__ gxy_c, const float* __restrict__ gxz_c, const float* __restrict__ gyz_c,
    const float* __restrict__ gxy_f, const float* __restrict__ gxz_f, const float* __restrict__ gyz_f,
    const float* __restrict__ cxy,   const float* __restrict__ cxz,   const float* __restrict__ cyz,
    const _Float16* __restrict__ wf16,
    const float* __restrict__ lin0_b, const float* __restrict__ lin1_b,
    const float* __restrict__ comb0_b, const float* __restrict__ comb1_b,
    const float* __restrict__ out_w, const float* __restrict__ out_b,
    const float* __restrict__ clin0_b, const float* __restrict__ clin1_b,
    const float* __restrict__ cout_w, const float* __restrict__ cout_b,
    float* __restrict__ out, int N) {
    // 4 waves * 2 ping-pong buffers * 16 rows * S_LDS halves = 34816 bytes
    __shared__ _Float16 smem[4 * 2 * 16 * S_LDS];

    const int tid    = threadIdx.x;
    const int wave   = tid >> 5;
    const int lane   = tid & 31;
    const int prow   = lane & 15;        // point row in the 16-point tile
    const int chbase = (lane >> 4) * 16; // which half of the 32 channels this lane samples

    _Float16* bufA = smem + wave * (2 * 16 * S_LDS);
    _Float16* bufB = bufA + 16 * S_LDS;
    _Float16* rowA = bufA + prow * S_LDS;
    _Float16* rowB = bufB + prow * S_LDS;

    const int base = blockIdx.x * 64 + wave * 16;
    const int idx  = base + prow;
    const int pidx = idx < N ? idx : N - 1;

    const float X = P[pidx * 3 + 0];
    const float Y = P[pidx * 3 + 1];
    const float Z = P[pidx * 3 + 2];

    // --- features: coarse -> bufA[:,0:32], fine -> bufA[:,64:96] (pre-placed for concat)
    sample_set(gxy_c, gxz_c, gyz_c, 128, X, Y, Z, chbase, rowA);
    sample_set(gxy_f, gxz_f, gyz_f, 512, X, Y, Z, chbase, rowA + 64);

    // --- sdf trunk (all WMMA)
    mlp_layer<32, 64, true>(bufA, bufB, wf16 + OFF_LIN0, lin0_b, lane);   // h0  -> bufB[:,0:64]
    mlp_layer<64, 64, true>(bufB, bufA, wf16 + OFF_LIN1, lin1_b, lane);   // h1  -> bufA[:,0:64]; concat with feat_f => bufA[:,0:96]
    mlp_layer<96, 128, true>(bufA, bufB, wf16 + OFF_COMB0, comb0_b, lane); // g0 -> bufB[:,0:128]
    mlp_layer<128, 128, true>(bufB, bufA, wf16 + OFF_COMB1, comb1_b, lane);// g1 -> bufA[:,0:128]

    // --- sdf head: 128-dot on lanes 0..15
    float sdf = 0.0f;
    if (lane < 16) {
        float s = out_b[0];
        const _Float16* g = bufA + prow * S_LDS;
#pragma unroll 8
        for (int k = 0; k < 128; ++k) s = fmaf((float)g[k], out_w[k], s);
        sdf = s;
    }

    // --- color branch: cf -> bufB[:,0:32] (g0 dead), then two WMMA layers
    sample_set(cxy, cxz, cyz, 256, X, Y, Z, chbase, rowB);
    mlp_layer<32, 64, true>(bufB, bufA, wf16 + OFF_CLIN0, clin0_b, lane); // hc0 -> bufA (g1 dead)
    mlp_layer<64, 64, true>(bufA, bufB, wf16 + OFF_CLIN1, clin1_b, lane); // hc1 -> bufB

    // --- rgb head + store: out[idx] = {sigmoid(rgb0..2), sdf}
    if (lane < 16 && idx < N) {
        const _Float16* hc = bufB + prow * S_LDS;
#pragma unroll
        for (int j = 0; j < 3; ++j) {
            float r = cout_b[j];
            const float* wr = cout_w + j * 64;
#pragma unroll 8
            for (int k = 0; k < 64; ++k) r = fmaf((float)hc[k], wr[k], r);
            out[idx * 4 + j] = 1.0f / (1.0f + expf(-r));
        }
        out[idx * 4 + 3] = sdf;
    }
}

extern "C" void kernel_launch(void* const* d_in, const int* in_sizes, int n_in,
                              void* d_out, int out_size, void* d_ws, size_t ws_size,
                              hipStream_t stream) {
    const float* P     = (const float*)d_in[0];
    const float* gxy_c = (const float*)d_in[1];
    const float* gxz_c = (const float*)d_in[2];
    const float* gyz_c = (const float*)d_in[3];
    const float* gxy_f = (const float*)d_in[4];
    const float* gxz_f = (const float*)d_in[5];
    const float* gyz_f = (const float*)d_in[6];
    const float* cxy   = (const float*)d_in[7];
    const float* cxz   = (const float*)d_in[8];
    const float* cyz   = (const float*)d_in[9];
    const float* lin0_w  = (const float*)d_in[10];
    const float* lin0_b  = (const float*)d_in[11];
    const float* lin1_w  = (const float*)d_in[12];
    const float* lin1_b  = (const float*)d_in[13];
    const float* comb0_w = (const float*)d_in[14];
    const float* comb0_b = (const float*)d_in[15];
    const float* comb1_w = (const float*)d_in[16];
    const float* comb1_b = (const float*)d_in[17];
    const float* out_w   = (const float*)d_in[18];
    const float* out_b   = (const float*)d_in[19];
    const float* clin0_w = (const float*)d_in[20];
    const float* clin0_b = (const float*)d_in[21];
    const float* clin1_w = (const float*)d_in[22];
    const float* clin1_b = (const float*)d_in[23];
    const float* cout_w  = (const float*)d_in[24];
    const float* cout_b  = (const float*)d_in[25];

    _Float16* wf16 = (_Float16*)d_ws;
    const int N = in_sizes[0] / 3;

    // Prologue: one-time f32 -> f16 weight conversion into workspace (deterministic each call).
    struct { const float* src; int off; int n; } cv[6] = {
        { lin0_w,  OFF_LIN0,  64 * 32  },
        { lin1_w,  OFF_LIN1,  64 * 64  },
        { comb0_w, OFF_COMB0, 128 * 96 },
        { comb1_w, OFF_COMB1, 128 * 128},
        { clin0_w, OFF_CLIN0, 64 * 32  },
        { clin1_w, OFF_CLIN1, 64 * 64  },
    };
    for (int i = 0; i < 6; ++i) {
        int n = cv[i].n;
        cvt_f32_to_f16<<<(n + 255) / 256, 256, 0, stream>>>(cv[i].src, wf16 + cv[i].off, n);
    }

    // Main fused kernel: 64 points per block (4 waves x 16-point WMMA tiles).
    int grid = (N + 63) / 64;
    triplane_decoder_kernel<<<grid, 128, 0, stream>>>(
        P, gxy_c, gxz_c, gyz_c, gxy_f, gxz_f, gyz_f, cxy, cxz, cyz,
        wf16, lin0_b, lin1_b, comb0_b, comb1_b, out_w, out_b,
        clin0_b, clin1_b, cout_w, cout_b,
        (float*)d_out, N);
}
